// GATAgent_22531398435302
// MI455X (gfx1250) — matrix operand order repaired and checked by reading, not compiled, not run on hardware
//
#include <hip/hip_runtime.h>

// Problem constants (match reference)
#define B_    8
#define N_    1024
#define INF_  1024
#define E_    256
#define H_    4
#define A_    64
#define HE_   (H_ * E_)   // 1024
#define BN_   (B_ * N_)   // 8192

typedef __attribute__((ext_vector_type(16))) __bf16 v16bf;
typedef __attribute__((ext_vector_type(8)))  __bf16 v8bf;
typedef __attribute__((ext_vector_type(8)))  float  v8f;

union ABfrag { v16bf v; v8bf q[2]; };

#define WMMA_BF16(Afr, Bfr, C) \
  C = __builtin_amdgcn_wmma_f32_16x16x32_bf16(false, (Afr).v, false, (Bfr).v, (short)0, C, false, false)

// ---------------- converts ----------------
__global__ void k_cvt_x(const float* __restrict__ x, __bf16* __restrict__ xb, int n) {
  int i = blockIdx.x * blockDim.x + threadIdx.x;
  if (i < n) xb[i] = (__bf16)x[i];
}

// W (H, IN_F, E) fp32 -> WbT (H, E, IN_F) bf16  (column-major for B-fragments)
__global__ void k_cvt_W(const float* __restrict__ W, __bf16* __restrict__ WbT) {
  int idx = blockIdx.x * blockDim.x + threadIdx.x;
  if (idx >= H_ * E_ * INF_) return;
  int i = idx % INF_;
  int o = (idx / INF_) % E_;
  int h = idx / (INF_ * E_);
  WbT[idx] = (__bf16)W[((size_t)h * INF_ + i) * E_ + o];
}

// W_act (HE, A) fp32 -> WactT (A, HE) bf16
__global__ void k_cvt_Wact(const float* __restrict__ Wact, __bf16* __restrict__ WactT) {
  int idx = blockIdx.x * blockDim.x + threadIdx.x;
  if (idx >= A_ * HE_) return;
  int k = idx % HE_;
  int a = idx / HE_;
  WactT[idx] = (__bf16)Wact[(size_t)k * A_ + a];
}

// ------- GEMM 1: WhT[h][b][e][n] = (X @ W[h])^T, bf16 in / f32 acc, pipelined -------
__global__ __launch_bounds__(128)
void k_gemm_wh(const __bf16* __restrict__ Xb, const __bf16* __restrict__ WbT,
               __bf16* __restrict__ WhT) {
  const int h     = blockIdx.y;
  const int mBase = blockIdx.x * 32;           // 32 output rows per block
  const int wave  = threadIdx.x >> 5;
  const int lane  = threadIdx.x & 31;
  const int r  = lane & 15;
  const int hi = lane >> 4;
  const int colBase = wave * 64;               // E column slice per wave

  const __bf16* aRow0 = Xb + (size_t)(mBase + r) * INF_ + hi * 8;
  const __bf16* aRow1 = aRow0 + (size_t)16 * INF_;
  const __bf16* bBase = WbT + ((size_t)h * E_ + colBase + r) * INF_ + hi * 8;

  v8f acc00 = {}, acc01 = {}, acc02 = {}, acc03 = {};
  v8f acc10 = {}, acc11 = {}, acc12 = {}, acc13 = {};
  ABfrag A0[2], A1[2];      // [parity]
  ABfrag Bf[2][4];          // [parity][nn]

#define LOAD_AB(P, KT) do {                                        \
    A0[P].q[0] = *(const v8bf*)(aRow0 + (KT));                     \
    A0[P].q[1] = *(const v8bf*)(aRow0 + (KT) + 16);                \
    A1[P].q[0] = *(const v8bf*)(aRow1 + (KT));                     \
    A1[P].q[1] = *(const v8bf*)(aRow1 + (KT) + 16);                \
    _Pragma("unroll")                                              \
    for (int nn = 0; nn < 4; ++nn) {                               \
      const __bf16* p_ = bBase + (size_t)nn * 16 * INF_ + (KT);    \
      Bf[P][nn].q[0] = *(const v8bf*)(p_);                         \
      Bf[P][nn].q[1] = *(const v8bf*)(p_ + 16);                    \
    } } while (0)

#define MMA_ALL(P) do {                                            \
    WMMA_BF16(A0[P], Bf[P][0], acc00); WMMA_BF16(A1[P], Bf[P][0], acc10); \
    WMMA_BF16(A0[P], Bf[P][1], acc01); WMMA_BF16(A1[P], Bf[P][1], acc11); \
    WMMA_BF16(A0[P], Bf[P][2], acc02); WMMA_BF16(A1[P], Bf[P][2], acc12); \
    WMMA_BF16(A0[P], Bf[P][3], acc03); WMMA_BF16(A1[P], Bf[P][3], acc13); \
  } while (0)

  LOAD_AB(0, 0);
  for (int kt = 0; kt < INF_; kt += 64) {
    LOAD_AB(1, kt + 32);
    MMA_ALL(0);
    if (kt + 64 < INF_) LOAD_AB(0, kt + 64);
    MMA_ALL(1);
  }
#undef LOAD_AB
#undef MMA_ALL

  const int bIdx = mBase >> 10;        // / N_
  const int nIdx = mBase & (N_ - 1);
  // C rows (M = g + hi*8) run along the contiguous N axis of WhT -> one b128 per frag
#define STORE_ACC(ACC, MI, NN) do {                                              \
    int eCol = colBase + (NN) * 16 + r;                                          \
    __bf16* dst = WhT + (((size_t)h * B_ + bIdx) * E_ + eCol) * N_               \
                  + nIdx + (MI) * 16 + hi * 8;                                   \
    v8bf pk;                                                                     \
    _Pragma("unroll") for (int g = 0; g < 8; ++g) pk[g] = (__bf16)(ACC)[g];      \
    *(v8bf*)dst = pk; } while (0)
  STORE_ACC(acc00, 0, 0); STORE_ACC(acc01, 0, 1);
  STORE_ACC(acc02, 0, 2); STORE_ACC(acc03, 0, 3);
  STORE_ACC(acc10, 1, 0); STORE_ACC(acc11, 1, 1);
  STORE_ACC(acc12, 1, 2); STORE_ACC(acc13, 1, 3);
#undef STORE_ACC
}

// ---------------- f1/f2 = Wh . a1/a2 ----------------
__global__ void k_f12(const __bf16* __restrict__ WhT, const float* __restrict__ a1,
                      const float* __restrict__ a2, float* __restrict__ f1,
                      float* __restrict__ f2) {
  int idx = blockIdx.x * blockDim.x + threadIdx.x;   // h*B*N
  if (idx >= H_ * B_ * N_) return;
  int n  = idx & (N_ - 1);
  int hb = idx >> 10;            // h*B + b
  int h  = hb >> 3;              // B_ == 8
  const __bf16* base = WhT + (size_t)hb * E_ * N_ + n;
  float s1 = 0.f, s2 = 0.f;
  for (int e = 0; e < E_; ++e) {
    float v = (float)base[(size_t)e * N_];
    s1 += v * a1[h * E_ + e];
    s2 += v * a2[h * E_ + e];
  }
  f1[idx] = s1;
  f2[idx] = s2;
}

// ------- fused masked-softmax attention + P@V (flash style, online softmax) -------
__global__ __launch_bounds__(128)
void k_attn(const __bf16* __restrict__ WhT, const float* __restrict__ f1,
            const float* __restrict__ f2, const int* __restrict__ adj,
            __bf16* __restrict__ Hp) {
  const int h = blockIdx.z, b = blockIdx.y;
  const int rowBase = blockIdx.x * 16;
  const int wave = threadIdx.x >> 5;
  const int lane = threadIdx.x & 31;
  const int r  = lane & 15;
  const int hi = lane >> 4;
  const int eBase = wave * 64;

  const float f1v    = f1[((size_t)h * B_ + b) * N_ + rowBase + r];
  const float* f2row = f2 + ((size_t)h * B_ + b) * N_;
  const int* adjRow  = adj + ((size_t)b * N_ + rowBase + r) * N_;
  const __bf16* vCol = WhT + (((size_t)h * B_ + b) * E_ + eBase + r) * N_ + hi * 8;

  float m = -3.0e38f, lsum = 0.f;
  v8f acc0 = {}, acc1 = {}, acc2 = {}, acc3 = {};

  // operand double buffers for the score inputs (next tile prefetched 1 iter ahead)
  float f2b[2][16];
  int   adjb[2][16];
#pragma unroll
  for (int i = 0; i < 8; ++i) {
    f2b[0][i]     = f2row[hi * 8 + i];
    f2b[0][i + 8] = f2row[16 + hi * 8 + i];
    adjb[0][i]     = adjRow[hi * 8 + i];
    adjb[0][i + 8] = adjRow[16 + hi * 8 + i];
  }

#define ATTN_STEP(PC, PN, JT) do {                                               \
    /* V-fragment loads first: in flight during the softmax VALU chain */        \
    ABfrag bb0, bb1, bb2, bb3;                                                   \
    { const __bf16* p_ = vCol + (JT);                                            \
      bb0.q[0] = *(const v8bf*)(p_);                                             \
      bb0.q[1] = *(const v8bf*)(p_ + 16);                                        \
      bb1.q[0] = *(const v8bf*)(p_ + (size_t)16 * N_);                           \
      bb1.q[1] = *(const v8bf*)(p_ + (size_t)16 * N_ + 16);                      \
      bb2.q[0] = *(const v8bf*)(p_ + (size_t)32 * N_);                           \
      bb2.q[1] = *(const v8bf*)(p_ + (size_t)32 * N_ + 16);                      \
      bb3.q[0] = *(const v8bf*)(p_ + (size_t)48 * N_);                           \
      bb3.q[1] = *(const v8bf*)(p_ + (size_t)48 * N_ + 16); }                    \
    if ((JT) + 32 < N_) {                                                        \
      _Pragma("unroll") for (int i = 0; i < 8; ++i) {                            \
        f2b[PN][i]      = f2row[(JT) + 32 + hi * 8 + i];                         \
        f2b[PN][i + 8]  = f2row[(JT) + 48 + hi * 8 + i];                         \
        adjb[PN][i]     = adjRow[(JT) + 32 + hi * 8 + i];                        \
        adjb[PN][i + 8] = adjRow[(JT) + 48 + hi * 8 + i]; } }                    \
    float s[16];                                                                 \
    _Pragma("unroll") for (int i = 0; i < 16; ++i) {                             \
      float e0 = f1v + f2b[PC][i];                                               \
      e0 = e0 > 0.f ? e0 : 0.2f * e0;                /* leaky relu */            \
      s[i] = adjb[PC][i] > 0 ? e0 : -9.0e15f; }      /* mask */                  \
    float tmax = s[0];                                                           \
    _Pragma("unroll") for (int i = 1; i < 16; ++i) tmax = fmaxf(tmax, s[i]);     \
    tmax = fmaxf(tmax, __shfl_xor(tmax, 16, 32));                                \
    float mNew  = fmaxf(m, tmax);                                                \
    float alpha = __expf(m - mNew);                                              \
    ABfrag Pf; float psum = 0.f;                                                 \
    _Pragma("unroll") for (int i = 0; i < 16; ++i) {                             \
      float pv = __expf(s[i] - mNew); psum += pv; Pf.v[i] = (__bf16)pv; }        \
    psum += __shfl_xor(psum, 16, 32);                                            \
    lsum = lsum * alpha + psum; m = mNew;                                        \
    float asc[8];                                                                \
    _Pragma("unroll") for (int g = 0; g < 8; ++g)                                \
      asc[g] = __shfl(alpha, g + hi * 8, 32);                                    \
    _Pragma("unroll") for (int g = 0; g < 8; ++g) {                              \
      acc0[g] *= asc[g]; acc1[g] *= asc[g];                                      \
      acc2[g] *= asc[g]; acc3[g] *= asc[g]; }                                    \
    WMMA_BF16(Pf, bb0, acc0); WMMA_BF16(Pf, bb1, acc1);                          \
    WMMA_BF16(Pf, bb2, acc2); WMMA_BF16(Pf, bb3, acc3);                          \
  } while (0)

  for (int jt = 0; jt < N_; jt += 64) {
    ATTN_STEP(0, 1, jt);
    ATTN_STEP(1, 0, jt + 32);
  }
#undef ATTN_STEP

  // finalize: divide by row sum, elu; stage tile in LDS for coalesced b128 stores
  __shared__ __bf16 lds[16 * 256];
  float inv[8];
#pragma unroll
  for (int g = 0; g < 8; ++g) inv[g] = 1.f / __shfl(lsum, g + hi * 8, 32);
#define ELU_LDS(ACC, NN) do {                                                    \
    _Pragma("unroll") for (int g = 0; g < 8; ++g) {                              \
      float v = (ACC)[g] * inv[g];                                               \
      v = v > 0.f ? v : (__expf(v) - 1.f);                                       \
      lds[(g + hi * 8) * 256 + eBase + (NN) * 16 + r] = (__bf16)v; } } while (0)
  ELU_LDS(acc0, 0); ELU_LDS(acc1, 1); ELU_LDS(acc2, 2); ELU_LDS(acc3, 3);
#undef ELU_LDS
  __syncthreads();
  {
    int t   = threadIdx.x;          // 128 threads x 32 elems = 16x256 tile
    int row = t >> 3;
    int col = (t & 7) * 32;
    const __bf16* src = lds + row * 256 + col;
    __bf16* dst = Hp + ((size_t)(b * N_ + rowBase + row)) * HE_ + (size_t)h * E_ + col;
#pragma unroll
    for (int i = 0; i < 4; ++i)
      *(v8bf*)(dst + i * 8) = *(const v8bf*)(src + i * 8);
  }
}

// ------- GEMM 2 (computed transposed): out^T = W_act^T @ feat^T, + bias -------
// A = WactT (A-dim x HE, row-major), B = Hp rows as columns -> C rows land
// contiguous in out[node][a].
__global__ __launch_bounds__(128)
void k_gemm_out(const __bf16* __restrict__ Hp, const __bf16* __restrict__ WactT,
                const float* __restrict__ b_act, float* __restrict__ out) {
  const int nodeBase = blockIdx.x * 16;
  const int wave = threadIdx.x >> 5;
  const int lane = threadIdx.x & 31;
  const int r  = lane & 15;
  const int hi = lane >> 4;

  const __bf16* aPtr = WactT + (size_t)(wave * 16 + r) * HE_ + hi * 8;   // A row = out channel
  const __bf16* bPtr = Hp + (size_t)(nodeBase + r) * HE_ + hi * 8;       // B col = node

  v8f acc = {};
  ABfrag Af[2], Bf[2];
#define LOAD_O(P, KT) do {                              \
    Af[P].q[0] = *(const v8bf*)(aPtr + (KT));           \
    Af[P].q[1] = *(const v8bf*)(aPtr + (KT) + 16);      \
    Bf[P].q[0] = *(const v8bf*)(bPtr + (KT));           \
    Bf[P].q[1] = *(const v8bf*)(bPtr + (KT) + 16); } while (0)
  LOAD_O(0, 0);
  for (int kt = 0; kt < HE_; kt += 64) {
    LOAD_O(1, kt + 32);
    WMMA_BF16(Af[0], Bf[0], acc);
    if (kt + 64 < HE_) LOAD_O(0, kt + 64);
    WMMA_BF16(Af[1], Bf[1], acc);
  }
#undef LOAD_O

  // C: row = a-channel (wave*16 + hi*8 + g), col = node (nodeBase + r)
  float* o = out + (size_t)(nodeBase + r) * A_ + wave * 16 + hi * 8;
  const float* bi = b_act + wave * 16 + hi * 8;
#pragma unroll
  for (int g = 0; g < 8; ++g) o[g] = acc[g] + bi[g];
}

// ---------------- host launch ----------------
extern "C" void kernel_launch(void* const* d_in, const int* in_sizes, int n_in,
                              void* d_out, int out_size, void* d_ws, size_t ws_size,
                              hipStream_t stream) {
  const float* x    = (const float*)d_in[0];
  const int*   adj  = (const int*)d_in[1];
  const float* W    = (const float*)d_in[2];
  const float* a1   = (const float*)d_in[3];
  const float* a2   = (const float*)d_in[4];
  const float* Wact = (const float*)d_in[5];
  const float* bact = (const float*)d_in[6];
  float* out = (float*)d_out;

  char* ws = (char*)d_ws;
  size_t off = 0;
  auto take = [&](size_t bytes) -> void* {
    void* p = ws + off;
    off += (bytes + 255) & ~(size_t)255;
    return p;
  };
  __bf16* Xb    = (__bf16*)take((size_t)BN_ * INF_ * 2);        // 16 MB
  __bf16* WbT   = (__bf16*)take((size_t)H_ * E_ * INF_ * 2);    // 2 MB
  __bf16* WactT = (__bf16*)take((size_t)A_ * HE_ * 2);          // 128 KB
  __bf16* WhT   = (__bf16*)take((size_t)H_ * B_ * E_ * N_ * 2); // 16 MB
  float*  f1    = (float*)take((size_t)H_ * B_ * N_ * 4);
  float*  f2    = (float*)take((size_t)H_ * B_ * N_ * 4);
  __bf16* Hp    = (__bf16*)take((size_t)BN_ * HE_ * 2);         // 16 MB

  k_cvt_x   <<<(BN_ * INF_ + 255) / 256, 256, 0, stream>>>(x, Xb, BN_ * INF_);
  k_cvt_W   <<<(H_ * E_ * INF_ + 255) / 256, 256, 0, stream>>>(W, WbT);
  k_cvt_Wact<<<(A_ * HE_ + 255) / 256, 256, 0, stream>>>(Wact, WactT);

  k_gemm_wh <<<dim3(BN_ / 32, H_), 128, 0, stream>>>(Xb, WbT, WhT);
  k_f12     <<<(H_ * B_ * N_ + 255) / 256, 256, 0, stream>>>(WhT, a1, a2, f1, f2);
  k_attn    <<<dim3(N_ / 16, B_, H_), 128, 0, stream>>>(WhT, f1, f2, adj, Hp);
  k_gemm_out<<<BN_ / 16, 128, 0, stream>>>(Hp, WactT, bact, out);
}
